// MultiHeadedAttention_47605417509346
// MI455X (gfx1250) — compile-verified, matrix operand order checked by reading, not run on hardware
//
#include <hip/hip_runtime.h>
#include <hip/hip_bf16.h>
#include <math.h>

typedef __attribute__((ext_vector_type(16))) _Float16 v16h;
typedef __attribute__((ext_vector_type(8)))  float    v8f;

#define Bc  4
#define Tc  2048
#define Dc  1024
#define Hc  16
#define DHc 64
#define Mc  (Bc*Tc)   // 8192 rows

__device__ __forceinline__ v8f wmma_f16(v16h a, v16h b, v8f c) {
  return __builtin_amdgcn_wmma_f32_16x16x32_f16(false, a, false, b, (short)0, c, false, false);
}

// ---------------- f32 -> f16 convert (one-shot pre-pass) ----------------
__global__ __launch_bounds__(256)
void cvt_f16_kernel(const float* __restrict__ in, _Float16* __restrict__ out) {
  size_t i = ((size_t)blockIdx.x * 256 + threadIdx.x) * 8;
  float4 a = *(const float4*)(in + i);
  float4 b = *(const float4*)(in + i + 4);
  union { _Float16 h[8]; uint4 u; } t;
  t.h[0] = (_Float16)a.x; t.h[1] = (_Float16)a.y;
  t.h[2] = (_Float16)a.z; t.h[3] = (_Float16)a.w;
  t.h[4] = (_Float16)b.x; t.h[5] = (_Float16)b.y;
  t.h[6] = (_Float16)b.z; t.h[7] = (_Float16)b.w;
  *(uint4*)(out + i) = t.u;
}

// ---------------- mask bit-pack: int32 mask -> 1 bit/key ----------------
__global__ __launch_bounds__(256)
void pack_mask_kernel(const int* __restrict__ mask, unsigned* __restrict__ bits) {
  size_t gid = (size_t)blockIdx.x * 256 + threadIdx.x;
  bool bit = mask[gid] != 0;
#if __has_builtin(__builtin_amdgcn_ballot_w32)
  unsigned word = __builtin_amdgcn_ballot_w32(bit);
#else
  unsigned word = (unsigned)__ballot(bit);
#endif
  if ((threadIdx.x & 31) == 0) bits[gid >> 5] = word;
}

// ---------------- GEMM: Y = A(8192x1024,f16) * W^T(1024x1024,f16) ----------------
// Double-buffered async global->LDS staging (ASYNCcnt pipeline).
// MODE 0: f16 out, head-major [b,h,t,d]   (Q, K)
// MODE 1: f16 out, transposed  [b,h,d,t]  (V)
// MODE 2: f32 out, row-major + bias       (output projection)
#define TILE_BYTES 12288u           // 128 rows * 48 halfs * 2B (96B pitch, 32B-aligned frags)
template<int MODE>
__global__ __launch_bounds__(256)
void gemm_xwt(const _Float16* __restrict__ Ap, const _Float16* __restrict__ W,
              const float* __restrict__ bias, void* __restrict__ outp, float alpha) {
  extern __shared__ __attribute__((aligned(128))) char smem[];
  const int tid  = threadIdx.x;
  const int lane = tid & 31;
  const int wave = tid >> 5;
  const int bm = blockIdx.x * 128;
  const int bn = blockIdx.y * 128;
  const int wm = (wave >> 2) * 64;   // 2 wave-rows
  const int wn = (wave & 3) * 32;    // 4 wave-cols
  const int hi = lane >> 4;          // half-wave selects K 0..15 / 16..31
  const int ln = lane & 15;
  const _Float16* Ab = Ap + (size_t)bm * Dc;
  const _Float16* Wb = W  + (size_t)bn * Dc;

  // chunk mapping for async staging: thread covers chunks (tid, tid+256);
  // chunk -> row = ch>>2, 8-half column group = (ch&3)*8
  const int r0 = tid >> 2,         kk0 = (tid & 3) * 8;
  const int r1 = (tid + 256) >> 2, kk1 = ((tid + 256) & 3) * 8;

  auto issue_stage = [&](int k0, unsigned bufbase) {
    const _Float16* g;
    unsigned l;
    g = Ab + (size_t)r0 * Dc + k0 + kk0; l = bufbase + r0 * 96 + kk0 * 2;
    asm volatile("global_load_async_to_lds_b128 %0, %1, off" :: "v"(l), "v"(g) : "memory");
    g = Ab + (size_t)r1 * Dc + k0 + kk1; l = bufbase + r1 * 96 + kk1 * 2;
    asm volatile("global_load_async_to_lds_b128 %0, %1, off" :: "v"(l), "v"(g) : "memory");
    g = Wb + (size_t)r0 * Dc + k0 + kk0; l = bufbase + TILE_BYTES + r0 * 96 + kk0 * 2;
    asm volatile("global_load_async_to_lds_b128 %0, %1, off" :: "v"(l), "v"(g) : "memory");
    g = Wb + (size_t)r1 * Dc + k0 + kk1; l = bufbase + TILE_BYTES + r1 * 96 + kk1 * 2;
    asm volatile("global_load_async_to_lds_b128 %0, %1, off" :: "v"(l), "v"(g) : "memory");
  };

  v8f acc[4][2] = {};

  const int NS = Dc / 32;            // 32 K-panels
  issue_stage(0, 0u);                // prologue: stage 0 -> buffer 0
  for (int s = 0; s < NS; ++s) {
    const unsigned cur = (s & 1) ? 2u * TILE_BYTES : 0u;
    if (s + 1 < NS) {
      issue_stage((s + 1) * 32, ((s + 1) & 1) ? 2u * TILE_BYTES : 0u);
      asm volatile("s_wait_asynccnt 0x4" ::: "memory");   // stage s's 4 ops retired
    } else {
      asm volatile("s_wait_asynccnt 0x0" ::: "memory");
    }
    __syncthreads();

    v16h a[4], b[2];
    #pragma unroll
    for (int mi = 0; mi < 4; ++mi)
      a[mi] = *reinterpret_cast<const v16h*>(smem + cur + (wm + mi*16 + ln) * 96 + hi * 32);
    #pragma unroll
    for (int ni = 0; ni < 2; ++ni)
      b[ni] = *reinterpret_cast<const v16h*>(smem + cur + TILE_BYTES + (wn + ni*16 + ln) * 96 + hi * 32);
    #pragma unroll
    for (int mi = 0; mi < 4; ++mi)
      #pragma unroll
      for (int ni = 0; ni < 2; ++ni)
        acc[mi][ni] = wmma_f16(a[mi], b[ni], acc[mi][ni]);
    __syncthreads();                 // releases buffer cur for stage s+2's writes
  }

  #pragma unroll
  for (int mi = 0; mi < 4; ++mi)
  #pragma unroll
  for (int ni = 0; ni < 2; ++ni)
  #pragma unroll
  for (int j = 0; j < 8; ++j) {
    int row = bm + wm + mi*16 + j + hi*8;
    int col = bn + wn + ni*16 + ln;
    float v = acc[mi][ni][j] * alpha;
    if (MODE == 2) {
      ((float*)outp)[(size_t)row * Dc + col] = v + bias[col];
    } else {
      int bb = row >> 11, t = row & (Tc - 1);
      int h  = col >> 6,  d = col & (DHc - 1);
      _Float16* o = (_Float16*)outp;
      if (MODE == 0) o[(((size_t)(bb*Hc + h))*Tc + t)*DHc + d] = (_Float16)v;
      else           o[(((size_t)(bb*Hc + h))*DHc + d)*Tc + t] = (_Float16)v;
    }
  }
}

// ---------------- flash attention: 32 q-rows / WG, head loop, 64-key tiles ----------------
__global__ __launch_bounds__(256)
void flash_attn_kernel(const _Float16* __restrict__ Qh, const _Float16* __restrict__ Kh,
                       const _Float16* __restrict__ Vt, const unsigned* __restrict__ mbits,
                       _Float16* __restrict__ Zc) {
  __shared__ float Sbuf[32][68];                 // pad 68: per-lane column stores spread banks
  __shared__ alignas(32) _Float16 Pbuf[32][80];  // 160B rows: 32B-aligned v16h frags
  __shared__ float rowred[32][8];
  __shared__ float m_s[32], l_s[32], sc_s[32];
  __shared__ unsigned mw[32][2];

  const int tid = threadIdx.x, lane = tid & 31, wave = tid >> 5;
  const int mi = wave & 1, ni = wave >> 1;       // wave owns S/Z tile (mi, ni)
  const int hi = lane >> 4, ln = lane & 15;
  const int qbase = blockIdx.x * 32;
  const int b = blockIdx.y;

  for (int h = 0; h < Hc; ++h) {
    __syncthreads();
    if (tid < 32) { m_s[tid] = -1e30f; l_s[tid] = 0.f; }
    v8f zacc = {};
    const size_t headT = (size_t)(b*Hc + h) * Tc;

    v16h aq[2];
    #pragma unroll
    for (int kk = 0; kk < 2; ++kk)
      aq[kk] = *reinterpret_cast<const v16h*>(
          Qh + (headT + qbase + mi*16 + ln) * DHc + kk*32 + hi*16);

    for (int kt = 0; kt < Tc/64; ++kt) {
      const int kbase = kt * 64;
      __syncthreads();                                    // LDS reuse fence
      if (tid < 64) {
        int r = tid >> 1, w = tid & 1;
        mw[r][w] = mbits[((size_t)b*Tc + qbase + r) * (Tc/32) + kt*2 + w];
      }
      // S tile = Q * K^T  (1/sqrt(dh) folded into Q)
      v8f s = {};
      #pragma unroll
      for (int kk = 0; kk < 2; ++kk) {
        v16h bk = *reinterpret_cast<const v16h*>(
            Kh + (headT + kbase + ni*16 + ln) * DHc + kk*32 + hi*16);
        s = wmma_f16(aq[kk], bk, s);
      }
      __syncthreads();                                    // mask words ready
      const int colL = ni*16 + ln;
      #pragma unroll
      for (int j = 0; j < 8; ++j) {
        int r = mi*16 + j + hi*8;
        float sv = s[j];
        if (((mw[r][colL >> 5] >> (colL & 31)) & 1u) == 0u) sv = -1e9f;
        Sbuf[r][colL] = sv;
      }
      __syncthreads();                                    // S ready
      { // partial row max (8 threads/row)
        int r = tid >> 3, c8 = (tid & 7) * 8;
        float mx = Sbuf[r][c8];
        #pragma unroll
        for (int t = 1; t < 8; ++t) mx = fmaxf(mx, Sbuf[r][c8 + t]);
        rowred[r][tid & 7] = mx;
      }
      __syncthreads();
      if (tid < 32) {                                     // online-softmax state
        float mx = rowred[tid][0];
        #pragma unroll
        for (int c = 1; c < 8; ++c) mx = fmaxf(mx, rowred[tid][c]);
        float mo = m_s[tid], mn = fmaxf(mo, mx);
        sc_s[tid] = __expf(mo - mn);
        m_s[tid] = mn;
      }
      __syncthreads();
      { // exponentiate -> P (f16) + partial sums
        int r = tid >> 3, c8 = (tid & 7) * 8;
        float mn = m_s[r], sum = 0.f;
        #pragma unroll
        for (int t = 0; t < 8; ++t) {
          float p = __expf(Sbuf[r][c8 + t] - mn);
          Pbuf[r][c8 + t] = (_Float16)p;
          sum += p;
        }
        rowred[r][tid & 7] = sum;
      }
      __syncthreads();                                    // P + sums ready
      if (tid < 32) {
        float sum = 0.f;
        #pragma unroll
        for (int c = 0; c < 8; ++c) sum += rowred[tid][c];
        l_s[tid] = l_s[tid] * sc_s[tid] + sum;
      }
      // rescale running Z and accumulate P * V
      #pragma unroll
      for (int j = 0; j < 8; ++j) zacc[j] *= sc_s[mi*16 + j + hi*8];
      #pragma unroll
      for (int kk = 0; kk < 2; ++kk) {
        v16h ap = *reinterpret_cast<const v16h*>(&Pbuf[mi*16 + ln][kk*32 + hi*16]);
        v16h bv = *reinterpret_cast<const v16h*>(
            Vt + (((size_t)(b*Hc + h))*DHc + ni*16 + ln) * Tc + kbase + kk*32 + hi*16);
        zacc = wmma_f16(ap, bv, zacc);
      }
    }
    __syncthreads();                                      // final l_s visible
    #pragma unroll
    for (int j = 0; j < 8; ++j) {
      int r = mi*16 + j + hi*8;
      float z = zacc[j] / l_s[r];
      int t = qbase + r, col = h*DHc + ni*16 + ln;
      Zc[((size_t)b*Tc + t) * Dc + col] = (_Float16)z;
    }
  }
}

extern "C" void kernel_launch(void* const* d_in, const int* in_sizes, int n_in,
                              void* d_out, int out_size, void* d_ws, size_t ws_size,
                              hipStream_t stream) {
  const float* q   = (const float*)d_in[0];
  const float* k   = (const float*)d_in[1];
  const float* v   = (const float*)d_in[2];
  const int*   msk = (const int*)d_in[3];
  const float* Wq  = (const float*)d_in[4];
  const float* Wk  = (const float*)d_in[5];
  const float* Wv  = (const float*)d_in[6];
  const float* Wo  = (const float*)d_in[7];
  const float* bo  = (const float*)d_in[8];
  float* out = (float*)d_out;

  char* ws = (char*)d_ws;
  const size_t MB   = 1u << 20;
  const size_t szAct = (size_t)Mc * Dc * sizeof(_Float16);   // 16 MB
  const size_t szW   = (size_t)Dc * Dc * sizeof(_Float16);   //  2 MB
  _Float16* xq16 = (_Float16*)(ws + 0*szAct);   // reused as Zc after projections
  _Float16* xk16 = (_Float16*)(ws + 1*szAct);
  _Float16* xv16 = (_Float16*)(ws + 2*szAct);
  _Float16* Qh   = (_Float16*)(ws + 3*szAct);
  _Float16* Kh   = (_Float16*)(ws + 4*szAct);
  _Float16* Vt   = (_Float16*)(ws + 5*szAct);
  _Float16* Wq16 = (_Float16*)(ws + 6*szAct);
  _Float16* Wk16 = (_Float16*)(ws + 6*szAct + 1*szW);
  _Float16* Wv16 = (_Float16*)(ws + 6*szAct + 2*szW);
  _Float16* Wo16 = (_Float16*)(ws + 6*szAct + 3*szW);
  unsigned* mb   = (unsigned*)(ws + 6*szAct + 4*szW);        //  2 MB bitmask
  _Float16* Zc   = xq16;

  const int actBlocks = (int)((size_t)Mc * Dc / (8 * 256)); // 4096
  const int wBlocks   = (int)((size_t)Dc * Dc / (8 * 256)); //  512
  hipLaunchKernelGGL(cvt_f16_kernel, dim3(actBlocks), dim3(256), 0, stream, q, xq16);
  hipLaunchKernelGGL(cvt_f16_kernel, dim3(actBlocks), dim3(256), 0, stream, k, xk16);
  hipLaunchKernelGGL(cvt_f16_kernel, dim3(actBlocks), dim3(256), 0, stream, v, xv16);
  hipLaunchKernelGGL(cvt_f16_kernel, dim3(wBlocks),   dim3(256), 0, stream, Wq, Wq16);
  hipLaunchKernelGGL(cvt_f16_kernel, dim3(wBlocks),   dim3(256), 0, stream, Wk, Wk16);
  hipLaunchKernelGGL(cvt_f16_kernel, dim3(wBlocks),   dim3(256), 0, stream, Wv, Wv16);
  hipLaunchKernelGGL(cvt_f16_kernel, dim3(wBlocks),   dim3(256), 0, stream, Wo, Wo16);

  hipLaunchKernelGGL(pack_mask_kernel, dim3((Bc*Tc*Tc)/256), dim3(256), 0, stream, msk, mb);

  dim3 gg(Mc/128, Dc/128);
  const size_t gemmLds = 4 * TILE_BYTES;   // 48 KB: 2 matrices x 2 buffers
  hipLaunchKernelGGL((gemm_xwt<0>), gg, dim3(256), gemmLds, stream,
                     xq16, Wq16, (const float*)nullptr, (void*)Qh, 0.125f); // 1/sqrt(64)
  hipLaunchKernelGGL((gemm_xwt<0>), gg, dim3(256), gemmLds, stream,
                     xk16, Wk16, (const float*)nullptr, (void*)Kh, 1.0f);
  hipLaunchKernelGGL((gemm_xwt<1>), gg, dim3(256), gemmLds, stream,
                     xv16, Wv16, (const float*)nullptr, (void*)Vt, 1.0f);

  hipLaunchKernelGGL(flash_attn_kernel, dim3(Tc/32, Bc), dim3(256), 0, stream,
                     Qh, Kh, Vt, mb, Zc);

  hipLaunchKernelGGL((gemm_xwt<2>), gg, dim3(256), gemmLds, stream,
                     Zc, Wo16, bo, (void*)out, 1.0f);
}
